// OptimizedMAGECActor_55267639165123
// MI455X (gfx1250) — matrix operand order repaired and checked by reading, not compiled, not run on hardware
//
#include <hip/hip_runtime.h>
#include <stdint.h>

typedef uint16_t u16;
typedef __attribute__((ext_vector_type(16))) __bf16 bf16x16;
typedef __attribute__((ext_vector_type(8)))  float  f32x8;
typedef __attribute__((ext_vector_type(4)))  unsigned int u32x4;
typedef __attribute__((address_space(3))) unsigned char lds_u8;

__device__ __forceinline__ u16 f2bf(float f){
  union { float f; unsigned u; } c; c.f = f;
  unsigned r = c.u + 0x7FFFu + ((c.u >> 16) & 1u);
  return (u16)(r >> 16);
}
__device__ __forceinline__ float bf2f(u16 h){
  union { unsigned u; float f; } c; c.u = ((unsigned)h) << 16;
  return c.f;
}

union FragAB { u32x4 q[2]; bf16x16 v; };

// ---------------- utility ----------------
__global__ void zero_f32(float* __restrict__ p, size_t n){
  size_t i = (size_t)blockIdx.x * blockDim.x + threadIdx.x;
  size_t st = (size_t)gridDim.x * blockDim.x;
  for (; i < n; i += st) p[i] = 0.0f;
}

// ---------------- input projection: h0 = bf16(x @ W_in + b_in) ----------------
__global__ void input_proj(const float* __restrict__ x, const float* __restrict__ Win,
                           const float* __restrict__ bin, u16* __restrict__ h0){
  const int i = blockIdx.x, c = threadIdx.x;
  float v = bin[c];
  #pragma unroll
  for (int f = 0; f < 4; ++f) v += x[(size_t)i*4 + f] * Win[f*128 + c];
  h0[(size_t)i*128 + c] = f2bf(v);
}

// ---------------- degree + edge-attr scatter (once; layer-invariant) ----------------
__global__ void deg_eagg(const int* __restrict__ dst, const float* __restrict__ eattr,
                         float* __restrict__ cnt, float* __restrict__ agge, int E){
  int e = blockIdx.x * blockDim.x + threadIdx.x;
  if (e >= E) return;
  int d = dst[e];
  atomicAdd(&cnt[d], 1.0f);
  atomicAdd(&agge[2*d + 0], eattr[2*e + 0]);
  atomicAdd(&agge[2*d + 1], eattr[2*e + 1]);
}

// ---------------- per-layer scatter: agg[dst] += h[src]  (wave per edge, 4 ch/lane) ----------------
__global__ __launch_bounds__(256)
void scatter_h(const u16* __restrict__ h, int ldh,
               const int* __restrict__ src, const int* __restrict__ dst,
               float* __restrict__ agg, int E){
  int wid = (int)blockIdx.x * (blockDim.x >> 5) + (threadIdx.x >> 5);
  if (wid >= E) return;
  int lane = threadIdx.x & 31;
  int s = src[wid], d = dst[wid];
  const uint2 pk = *(const uint2*)(h + (size_t)s*ldh + lane*4);
  float* ap = agg + (size_t)d*128 + lane*4;
  atomicAdd(ap+0, bf2f((u16)(pk.x & 0xFFFFu)));
  atomicAdd(ap+1, bf2f((u16)(pk.x >> 16)));
  atomicAdd(ap+2, bf2f((u16)(pk.y & 0xFFFFu)));
  atomicAdd(ap+3, bf2f((u16)(pk.y >> 16)));
}

// ---------------- build fused A = [ (agg+h)/cnt | h ] (bf16, K=256) + e01 ----------------
__global__ void build_a(const u16* __restrict__ h, int ldh, const float* __restrict__ agg,
                        const float* __restrict__ agge, const float* __restrict__ cnt,
                        u16* __restrict__ A, float* __restrict__ e01){
  const int i = blockIdx.x, c = threadIdx.x;
  const float inv = 1.0f / (cnt[i] + 1.0f);     // self loop -> deg+1 >= 1
  const u16 hb = h[(size_t)i*ldh + c];
  const float a = (agg[(size_t)i*128 + c] + bf2f(hb)) * inv;
  A[(size_t)i*256 + c]       = f2bf(a);
  A[(size_t)i*256 + 128 + c] = hb;
  if (c < 2) e01[2*i + c] = agge[2*i + c] * inv;
}

// ---------------- async stage of one packed-B k-slice into LDS ----------------
template<int NT>
__device__ __forceinline__ void stage_async(const u16* __restrict__ Wp,
                                            unsigned char* sb, int tid, int ks){
  const int CH = (NT * 32 * 32) / 16;   // 16B chunks per k-slice
  const unsigned char* g = (const unsigned char*)(Wp + (size_t)ks * NT * 32 * 16);
  for (int c = tid; c < CH; c += 128){
    lds_u8* lp = (lds_u8*)(sb + (size_t)c * 16);
    asm volatile("global_load_async_to_lds_b128 %0, %1, off"
                 :: "v"(lp), "v"(g + (size_t)c * 16) : "memory");
  }
}

// ---------------- WMMA GEMM: C[N,16*NT] = A[N,K](bf16) @ Wpack, B via async->LDS ----------------
// EPI 0: layer (+= e0*WnRow128 + e1*WnRow129, relu, f32 out)
// EPI 1: jump  (+bias, relu, bf16 out)
// EPI 2: score (+bias, relu, f32 out)
template<int K, int NT, int EPI>
__global__ __launch_bounds__(128)
void gemm_wmma(const u16* __restrict__ A, int lda, const u16* __restrict__ Wp,
               const float* __restrict__ epA, const float* __restrict__ epB,
               const float* __restrict__ e01,
               float* __restrict__ outF, u16* __restrict__ outB, int ldo, int Nrows)
{
  constexpr int KS = K / 32;
  __shared__ unsigned char sbuf[2][NT * 32 * 32];   // double-buffered B k-slice

  const int tid  = threadIdx.x;
  const int lane = tid & 31;
  const int wave = (int)blockIdx.x * 4 + (tid >> 5);
  const int Mtiles = (Nrows + 15) >> 4;
  const bool active = (wave < Mtiles);              // tail waves still stage + barrier
  const int m0 = (active ? wave : 0) << 4;
  const int half = lane >> 4;
  int mA = m0 + (lane & 15);
  if (mA >= Nrows) mA = Nrows - 1;                  // clamp A reads; EXEC stays all-ones
  const u16* Arow = A + (size_t)mA * lda;

  stage_async<NT>(Wp, sbuf[0], tid, 0);             // prologue stage

  f32x8 acc[NT];
  #pragma unroll
  for (int t = 0; t < NT; ++t) acc[t] = (f32x8)(0.0f);

  #pragma unroll
  for (int ks = 0; ks < KS; ++ks){
    asm volatile("s_wait_asynccnt 0x0" ::: "memory"); // my async stores landed
    __syncthreads();                                  // everyone's landed
    if (ks + 1 < KS) stage_async<NT>(Wp, sbuf[(ks + 1) & 1], tid, ks + 1);

    const int kk = ks * 32;
    FragAB fa;                                        // A: contiguous k = 16*half + s (matches packed B)
    fa.q[0] = *(const u32x4*)(Arow + kk + half*16);
    fa.q[1] = *(const u32x4*)(Arow + kk + half*16 + 8);
    const u32x4* bs = (const u32x4*)sbuf[ks & 1];
    #pragma unroll
    for (int t = 0; t < NT; ++t){
      FragAB fb;                                      // ds_load_b128 x2, 32B/lane
      fb.q[0] = bs[(t*32 + lane)*2 + 0];
      fb.q[1] = bs[(t*32 + lane)*2 + 1];
      acc[t] = __builtin_amdgcn_wmma_f32_16x16x32_bf16(
                   false, fa.v, false, fb.v, (short)0, acc[t], false, false);
    }
    __syncthreads();                                  // reads done before buffer reuse
  }

  if (!active) return;
  #pragma unroll
  for (int t = 0; t < NT; ++t){
    const int n = t*16 + (lane & 15);
    #pragma unroll
    for (int r = 0; r < 8; ++r){
      const int m = m0 + half*8 + r;                  // C/D layout: VGPR r -> row r + half*8
      if (m >= Nrows) continue;
      float v = acc[t][r];
      if (EPI == 0){
        v += e01[2*m] * epA[n] + e01[2*m+1] * epB[n]; // eattr rank-2 update
        outF[(size_t)m*ldo + n] = fmaxf(v, 0.0f);
      } else if (EPI == 1){
        outB[(size_t)m*ldo + n] = f2bf(fmaxf(v + epA[n], 0.0f));
      } else {
        outF[(size_t)m*ldo + n] = fmaxf(v + epA[n], 0.0f);
      }
    }
  }
}

// ---------------- weight repack into per-lane B-fragment layout ----------------
__global__ void pack_w(const float* __restrict__ W, u16* __restrict__ Wp, int K, int Ncols){
  const int NT = Ncols >> 4;
  const int total = (K >> 5) * NT * 32;
  int t = blockIdx.x * blockDim.x + threadIdx.x;
  if (t >= total) return;
  int lane = t & 31, nt = (t >> 5) % NT, ks = t / (32 * NT);
  int n = nt*16 + (lane & 15), kb = ks*32 + (lane >> 4)*16;
  u16* dstp = Wp + (size_t)t * 16;
  for (int j = 0; j < 16; ++j) dstp[j] = f2bf(W[(size_t)(kb + j)*Ncols + n]);
}

__global__ void pack_wcat(const float* __restrict__ Wn, const float* __restrict__ Ws,
                          u16* __restrict__ Wp){      // K=256, Ncols=128
  int t = blockIdx.x * blockDim.x + threadIdx.x;
  if (t >= 8*8*32) return;
  int lane = t & 31, nt = (t >> 5) & 7, ks = t >> 8;
  int n = nt*16 + (lane & 15), kb = ks*32 + (lane >> 4)*16;
  u16* dstp = Wp + (size_t)t * 16;
  for (int j = 0; j < 16; ++j){
    int k = kb + j;
    float w = (k < 128) ? Wn[(size_t)k*128 + n] : Ws[(size_t)(k-128)*128 + n];
    dstp[j] = f2bf(w);
  }
}

// ---------------- BatchNorm stats / finalize ----------------
__global__ void bn_stats(const float* __restrict__ out, float* __restrict__ sum,
                         float* __restrict__ sumsq, int N){
  const int c = threadIdx.x;
  size_t r0 = (size_t)blockIdx.x * 256, rend = r0 + 256;
  if (rend > (size_t)N) rend = N;
  float s = 0.f, q = 0.f;
  for (size_t r = r0; r < rend; ++r){ float v = out[r*128 + c]; s += v; q += v*v; }
  atomicAdd(&sum[c], s);
  atomicAdd(&sumsq[c], q);
}

__global__ void bn_final(const float* sum, const float* sumsq, const float* gamma,
                         const float* beta, float* scale, float* shift, int N){
  int c = threadIdx.x;
  float inv = 1.0f / (float)N;
  float mean = sum[c]*inv;
  float var  = sumsq[c]*inv - mean*mean;
  float sc   = gamma[c] * rsqrtf(var + 1e-5f);
  scale[c] = sc;
  shift[c] = beta[c] - mean*sc;
}

// ---------------- BN apply + L2 row-normalize -> bf16 jump slice ----------------
__global__ void bn_l2_norm(const float* __restrict__ out, const float* __restrict__ scale,
                           const float* __restrict__ shift, u16* __restrict__ dstB, int ldd){
  const int i = blockIdx.x, c = threadIdx.x;
  float v = out[(size_t)i*128 + c]*scale[c] + shift[c];
  __shared__ float red[128];
  red[c] = v*v; __syncthreads();
  for (int s = 64; s > 0; s >>= 1){ if (c < s) red[c] += red[c + s]; __syncthreads(); }
  float nrm = fmaxf(sqrtf(red[0]), 1e-12f);
  dstB[(size_t)i*ldd + c] = f2bf(v / nrm);
}

// ---------------- final scorer GEMV ----------------
__global__ void score_k(const float* __restrict__ hidden, const float* __restrict__ W2,
                        const float* __restrict__ b2, float* __restrict__ out, int N){
  int i = blockIdx.x * blockDim.x + threadIdx.x;
  if (i >= N) return;
  float s = b2[0];
  #pragma unroll 8
  for (int k = 0; k < 64; ++k) s += hidden[(size_t)i*64 + k] * W2[k];
  out[i] = s;
}

extern "C" void kernel_launch(void* const* d_in, const int* in_sizes, int n_in,
                              void* d_out, int out_size, void* d_ws, size_t ws_size,
                              hipStream_t stream) {
  const float* x      = (const float*)d_in[0];
  const int*   eidx   = (const int*)  d_in[1];
  const float* eattr  = (const float*)d_in[2];
  const float* W_in   = (const float*)d_in[3];
  const float* b_in   = (const float*)d_in[4];
  const float* W_jump = (const float*)d_in[5];
  const float* b_jump = (const float*)d_in[6];
  const float* W1     = (const float*)d_in[7];
  const float* b1     = (const float*)d_in[8];
  const float* W2     = (const float*)d_in[9];
  const float* b2     = (const float*)d_in[10];
  const float* Wn[3]  = {(const float*)d_in[11], (const float*)d_in[15], (const float*)d_in[19]};
  const float* Wsl[3] = {(const float*)d_in[12], (const float*)d_in[16], (const float*)d_in[20]};
  const float* gam[3] = {(const float*)d_in[13], (const float*)d_in[17], (const float*)d_in[21]};
  const float* bet[3] = {(const float*)d_in[14], (const float*)d_in[18], (const float*)d_in[22]};

  const int N = in_sizes[0] / 4;
  const int E = in_sizes[2] / 2;
  const int* src = eidx;
  const int* dst = eidx + E;

  char* ws = (char*)d_ws; size_t off = 0;
  auto carve = [&](size_t bytes)->char* {
    char* p = ws + off; off = (off + bytes + 511) & ~(size_t)511; return p;
  };
  u16*   A      = (u16*)  carve((size_t)N*256*2);   // fused GEMM A
  float* X      = (float*)carve((size_t)N*128*4);   // agg (scatter) then layer out (aliased)
  u16*   jumpB  = (u16*)  carve((size_t)N*384*2);   // concat of 3 layer outputs (bf16)
  u16*   h0     = (u16*)  carve((size_t)N*128*2);   // input projection
  u16*   jout   = (u16*)  carve((size_t)N*128*2);   // jump output (bf16 A for scorer)
  float* hidden = (float*)carve((size_t)N*64*4);
  float* cnt    = (float*)carve((size_t)N*4);
  float* agge   = (float*)carve((size_t)N*8);
  float* e01    = (float*)carve((size_t)N*8);
  float* bnsum  = (float*)carve(128*4);
  float* bnsq   = (float*)carve(128*4);
  float* bnsc   = (float*)carve(128*4);
  float* bnsh   = (float*)carve(128*4);
  u16* WcatP[3]; for (int l = 0; l < 3; ++l) WcatP[l] = (u16*)carve(256*128*2);
  u16* WjP = (u16*)carve(384*128*2);
  u16* W1P = (u16*)carve(128*64*2);
  (void)ws_size; (void)n_in; (void)out_size;

  // one-time per launch: degree + eattr scatter, input projection, weight packs
  zero_f32<<<1024, 256, 0, stream>>>(cnt, (size_t)N);
  zero_f32<<<1024, 256, 0, stream>>>(agge, (size_t)2*N);
  deg_eagg<<<(E + 255)/256, 256, 0, stream>>>(dst, eattr, cnt, agge, E);
  input_proj<<<N, 128, 0, stream>>>(x, W_in, b_in, h0);
  for (int l = 0; l < 3; ++l)
    pack_wcat<<<8, 256, 0, stream>>>(Wn[l], Wsl[l], WcatP[l]);
  pack_w<<<12, 256, 0, stream>>>(W_jump, WjP, 384, 128);
  pack_w<<<2, 256, 0, stream>>>(W1, W1P, 128, 64);

  const int Mtiles  = (N + 15) / 16;
  const int gblocks = (Mtiles + 3) / 4;     // 4 waves (wave32) per 128-thread block

  for (int l = 0; l < 3; ++l){
    const u16* h = (l == 0) ? h0 : (jumpB + (size_t)(l-1)*128);
    const int ldh = (l == 0) ? 128 : 384;
    zero_f32<<<8192, 256, 0, stream>>>(X, (size_t)N*128);
    scatter_h<<<(E + 7)/8, 256, 0, stream>>>(h, ldh, src, dst, X, E);
    build_a<<<N, 128, 0, stream>>>(h, ldh, X, agge, cnt, A, e01);
    gemm_wmma<256, 8, 0><<<gblocks, 128, 0, stream>>>(
        A, 256, WcatP[l], Wn[l] + 128*128, Wn[l] + 129*128, e01, X, nullptr, 128, N);
    zero_f32<<<1, 128, 0, stream>>>(bnsum, 128);
    zero_f32<<<1, 128, 0, stream>>>(bnsq, 128);
    bn_stats<<<(N + 255)/256, 128, 0, stream>>>(X, bnsum, bnsq, N);
    bn_final<<<1, 128, 0, stream>>>(bnsum, bnsq, gam[l], bet[l], bnsc, bnsh, N);
    bn_l2_norm<<<N, 128, 0, stream>>>(X, bnsc, bnsh, jumpB + (size_t)l*128, 384);
  }

  gemm_wmma<384, 8, 1><<<gblocks, 128, 0, stream>>>(
      jumpB, 384, WjP, b_jump, nullptr, nullptr, nullptr, jout, 128, N);
  gemm_wmma<128, 4, 2><<<gblocks, 128, 0, stream>>>(
      jout, 128, W1P, b1, nullptr, nullptr, hidden, nullptr, 64, N);
  score_k<<<(N + 255)/256, 256, 0, stream>>>(hidden, W2, b2, (float*)d_out, N);
}
